// CriticHead_13417477833493
// MI455X (gfx1250) — matrix operand order repaired and checked by reading, not compiled, not run on hardware
//
#include <hip/hip_runtime.h>
#include <hip/hip_bf16.h>
#include <math.h>

// ---------------------------------------------------------------------------
// CriticHead fused implementation for gfx1250 (MI455X).
//   y2 = relu([outer(res,fr,estep)(150) || backbone(768)] @ W1 + b1)   [T,128]
//   y41 = meanE * y2 ; y42 = sumC * y2      (per-task scalars)
//   out = (1-p)*FAILC + p*(y61+y62), p = sig(m*d3+b3)*sig(s*d5+b5)
// GEMM runs on V_WMMA_F32_16X16X32_BF16 (bf16 in, f32 accumulate).
// ---------------------------------------------------------------------------

typedef __attribute__((ext_vector_type(16))) __bf16 v16bf;
typedef __attribute__((ext_vector_type(8)))  __bf16 v8bf;
typedef __attribute__((ext_vector_type(8)))  float  v8f;

#define D_BB   768
#define D_IN   918
#define KPAD   928        // 29 * 32
#define KCH    29
#define D_H    128
#define NT     8          // 128 / 16 column tiles
#define FAILC  (-100.0f)

__device__ __forceinline__ unsigned short f2bf(float f) {
  unsigned int u = __float_as_uint(f);
  u += 0x7FFFu + ((u >> 16) & 1u);   // round-to-nearest-even
  return (unsigned short)(u >> 16);
}

// --- Kernel 1: build bf16 feature rows + per-task scalars -------------------
__global__ void critic_prep_kernel(const float* __restrict__ bb,
                                   const float* __restrict__ res,
                                   const float* __restrict__ fr,
                                   const float* __restrict__ estep,
                                   const float* __restrict__ enode,
                                   const float* __restrict__ ccl,
                                   const float* __restrict__ cnd,
                                   unsigned short* __restrict__ X,
                                   float* __restrict__ meanE,
                                   float* __restrict__ sumC) {
  const int j   = blockIdx.x;
  const int tid = threadIdx.x;
  unsigned short* Xr = X + (size_t)j * KPAD;

  if (tid < 150) {                     // einsum 'jn,jm,jo->jnmo' flatten: n*30+m*6+o
    int n = tid / 30, r = tid % 30, m = r / 6, o = r % 6;
    Xr[tid] = f2bf(res[j * 5 + n] * fr[j * 5 + m] * estep[j * 6 + o]);
  }
  for (int i = tid; i < D_BB; i += blockDim.x)
    Xr[150 + i] = f2bf(bb[(size_t)j * D_BB + i]);
  if (tid >= 150 && tid < 160)         // zero-pad K 918..927
    Xr[D_IN + (tid - 150)] = 0;

  if (tid == 160) {
    float s = 0.f;
    for (int i = 0; i < 64; ++i) s += enode[j * 64 + i];
    meanE[j] = s * (1.0f / 64.0f);
  }
  if (tid == 161) {
    float a = 0.f, b = 0.f;
    for (int i = 0; i < 4;  ++i) a += ccl[j * 4  + i];
    for (int i = 0; i < 32; ++i) b += cnd[j * 32 + i];
    sumC[j] = a * b;
  }
}

// --- Kernel 2: pack W1 into WMMA B-operand (32x16 bf16) lane layout ---------
// Block b = kc*8 + nt. Lane L: N = nt*16 + (L&15), element e -> K = kc*32 + (L>>4)*16 + e.
__global__ void critic_packW1_kernel(const float* __restrict__ W1,
                                     unsigned short* __restrict__ Bp) {
  const int b  = blockIdx.x;           // 0..231
  const int kc = b >> 3, nt = b & 7;
  const int L  = threadIdx.x;          // 32 lanes
  const int n  = nt * 16 + (L & 15);
  const int kb = kc * 32 + (L >> 4) * 16;
  unsigned short* dst = Bp + (size_t)b * 512 + L * 16;
  #pragma unroll
  for (int e = 0; e < 16; ++e) {
    int k = kb + e;
    dst[e] = (k < D_IN) ? f2bf(W1[(size_t)k * D_H + n]) : (unsigned short)0;
  }
}

// --- Kernel 3: WMMA GEMM tile + fused epilogue ------------------------------
__global__ void __launch_bounds__(256)
critic_gemm_kernel(const unsigned short* __restrict__ X,
                   const unsigned short* __restrict__ Bp,
                   const float* __restrict__ b1,
                   const float* __restrict__ W3, const float* __restrict__ b3,
                   const float* __restrict__ W4, const float* __restrict__ b4,
                   const float* __restrict__ W5, const float* __restrict__ b5,
                   const float* __restrict__ W6, const float* __restrict__ b6,
                   const float* __restrict__ meanE,
                   const float* __restrict__ sumC,
                   float* __restrict__ out) {
  __shared__ float y2s[16 * 132];      // stride 132 avoids bank conflicts
  __shared__ float dots[16 * 4];

  const int tid  = threadIdx.x;
  const int w    = tid >> 5;           // wave id = N-tile 0..7
  const int L    = tid & 31;
  const int half = L >> 4;
  const int col  = L & 15;
  const int mbase = blockIdx.x * 16;

  const unsigned short* Xr = X + (size_t)(mbase + col) * KPAD; // A row per lane
  const unsigned short* Bw = Bp + (size_t)w * 512;

  v8f c = {};
  for (int kc = 0; kc < KCH; ++kc) {
    // A 16x32 bf16 layout (matches packed B): lanes 0-15 hold K [0..15], lanes 16-31 [16..31]
    const unsigned short* ap = Xr + kc * 32 + half * 16;
    v8bf alo = *(const v8bf*)(ap);
    v8bf ahi = *(const v8bf*)(ap + 8);
    v16bf a;
    #pragma unroll
    for (int e = 0; e < 8; ++e) { a[e] = alo[e]; a[e + 8] = ahi[e]; }

    v16bf bm = *(const v16bf*)(Bw + (size_t)kc * NT * 512 + L * 16);

    c = __builtin_amdgcn_wmma_f32_16x16x32_bf16(false, a, false, bm,
                                                (short)0, c, false, false);
  }

  // bias + relu, stash y2 tile to LDS (C layout: VGPR v -> row v + 8*half, col = L&15)
  const float bias = b1[w * 16 + col];
  #pragma unroll
  for (int v = 0; v < 8; ++v) {
    float y = fmaxf(c[v] + bias, 0.0f);
    y2s[(v + 8 * half) * 132 + (w * 16 + col)] = y;
  }
  __syncthreads();

  // four length-128 head dot products per row
  if (tid < 64) {
    const int r = tid >> 2, h = tid & 3;
    const float* Wh = (h == 0) ? W3 : (h == 1) ? W4 : (h == 2) ? W5 : W6;
    float s = 0.f;
    #pragma unroll 4
    for (int k = 0; k < D_H; ++k) s += y2s[r * 132 + k] * Wh[k];
    dots[r * 4 + h] = s;
  }
  __syncthreads();

  if (tid < 16) {
    const int task = mbase + tid;
    const float m  = meanE[task];
    const float sc = sumC[task];
    const float d3 = dots[tid * 4 + 0], d4 = dots[tid * 4 + 1];
    const float d5 = dots[tid * 4 + 2], d6 = dots[tid * 4 + 3];
    const float y51 = 1.0f / (1.0f + __expf(-(m * d3 + b3[0])));
    const float y61 = m * d4 + b4[0];
    const float y52 = 1.0f / (1.0f + __expf(-(sc * d5 + b5[0])));
    const float y62 = sc * d6 + b6[0];
    const float p = y51 * y52;
    out[task] = (1.0f - p) * FAILC + p * (y61 + y62);
  }
}

// ---------------------------------------------------------------------------
extern "C" void kernel_launch(void* const* d_in, const int* in_sizes, int n_in,
                              void* d_out, int out_size, void* d_ws, size_t ws_size,
                              hipStream_t stream) {
  (void)n_in; (void)out_size; (void)ws_size;
  const float* bb    = (const float*)d_in[2];
  const float* res   = (const float*)d_in[3];
  const float* fr    = (const float*)d_in[4];
  const float* estep = (const float*)d_in[5];
  const float* enode = (const float*)d_in[6];
  const float* ccl   = (const float*)d_in[7];
  const float* cnd   = (const float*)d_in[8];
  const float* W1    = (const float*)d_in[9];
  const float* b1    = (const float*)d_in[10];
  const float* W3 = (const float*)d_in[11]; const float* b3 = (const float*)d_in[12];
  const float* W4 = (const float*)d_in[13]; const float* b4 = (const float*)d_in[14];
  const float* W5 = (const float*)d_in[15]; const float* b5 = (const float*)d_in[16];
  const float* W6 = (const float*)d_in[17]; const float* b6 = (const float*)d_in[18];
  float* out = (float*)d_out;

  const int T = in_sizes[2] / D_BB;    // 4096 for the reference shapes

  // workspace layout (256B-aligned slabs): X bf16 [T,928] | packed W1 | meanE | sumC
  char* ws = (char*)d_ws;
  const size_t xbytes  = (((size_t)T * KPAD * 2) + 255) & ~(size_t)255;
  const size_t bpbytes = (((size_t)KCH * NT * 512 * 2) + 255) & ~(size_t)255;
  unsigned short* X  = (unsigned short*)ws;
  unsigned short* Bp = (unsigned short*)(ws + xbytes);
  float* meanE = (float*)(ws + xbytes + bpbytes);
  float* sumC  = meanE + T;

  critic_prep_kernel<<<T, 256, 0, stream>>>(bb, res, fr, estep, enode, ccl, cnd,
                                            X, meanE, sumC);
  critic_packW1_kernel<<<KCH * NT, 32, 0, stream>>>(W1, Bp);
  critic_gemm_kernel<<<T / 16, 256, 0, stream>>>(X, Bp, b1, W3, b3, W4, b4,
                                                 W5, b5, W6, b6, meanE, sumC, out);
}
